// MultiheadAttention_23287312679473
// MI455X (gfx1250) — matrix operand order repaired and checked by reading, not compiled, not run on hardware
//
#include <hip/hip_runtime.h>

// ---------------------------------------------------------------------------
// Multi-head attention, MI455X (gfx1250), wave32, bf16 WMMA w/ f32 accumulate.
// B=2, L=2048, D=1024, H=16, d_head=64.
// Pipeline:
//   cvt kernels  : activations f32->bf16 rowmajor; weights f32->bf16 pair-packed
//   3x GEMM      : Q/K/V projections (64x128 tile, async-LDS double buffer)
//   flash attn   : 64-key tiles, TDM (tensor_load_to_lds) K/V staging,
//                  online softmax, bias in WMMA C, rowsum by WMMA
//   1x GEMM      : @Wo -> f32 output
// ---------------------------------------------------------------------------

typedef unsigned int uint32;
typedef unsigned long long uint64;
typedef __bf16 bf16;
typedef bf16  bf16x16 __attribute__((ext_vector_type(16)));
typedef float f32x8   __attribute__((ext_vector_type(8)));
typedef int   v4i     __attribute__((ext_vector_type(4)));
typedef unsigned int u32x4 __attribute__((ext_vector_type(4)));
typedef int   i32x4   __attribute__((ext_vector_type(4)));
typedef int   i32x8   __attribute__((ext_vector_type(8)));

union FragBF { bf16x16 v; uint32 u[8]; };
union FragF  { f32x8   v; float  f[8]; };

enum { L_SEQ = 2048, D_MODEL = 1024, N_HEADS = 16, D_HEAD = 64, BL = 4096 };

// CDNA5 WMMA 16-bit A-matrix (16x32) K mapping: VGPR i, lane-half h.
__device__ __forceinline__ int a_k0(int i, int half) {
    return ((i & 4) << 2) + ((i & 3) << 1) + (half << 3);
}
// CDNA5 WMMA 16-bit B-matrix (32x16): same K mapping as A (VGPR i, lane-half h)
__device__ __forceinline__ int b_k0(int i, int half) {
    return ((i & 4) << 2) + ((i & 3) << 1) + (half << 3);
}

__device__ __forceinline__ f32x8 wmma_bf16(const FragBF& a, const FragBF& b, f32x8 c) {
    return __builtin_amdgcn_wmma_f32_16x16x32_bf16(
        false, a.v, false, b.v, (short)0, c, false, false);
}

__device__ __forceinline__ uint32 pack_bf2(float a, float b) {
    union { bf16 h[2]; uint32 u; } r;
    r.h[0] = (bf16)a; r.h[1] = (bf16)b;
    return r.u;
}

// ---- CDNA5 async global->LDS copy (ASYNCcnt), with safe fallback ----------
#if __has_builtin(__builtin_amdgcn_global_load_async_to_lds_b128) && \
    __has_builtin(__builtin_amdgcn_s_wait_asynccnt)
#define HAVE_ASYNC_LDS 1
#else
#define HAVE_ASYNC_LDS 0
#endif

__device__ __forceinline__ void cp16(const void* g, void* l) {
#if HAVE_ASYNC_LDS
    __builtin_amdgcn_global_load_async_to_lds_b128(
        (__attribute__((address_space(1))) v4i*)g,
        (__attribute__((address_space(3))) v4i*)l, 0, 0);
#else
    *(uint4*)l = *(const uint4*)g;
#endif
}
__device__ __forceinline__ void wait_cp() {
#if HAVE_ASYNC_LDS
    __builtin_amdgcn_s_wait_asynccnt(0);
#endif
}

// ---- CDNA5 Tensor Data Mover (TENSORcnt), with async/sync fallback --------
#if __has_builtin(__builtin_amdgcn_tensor_load_to_lds) && \
    __has_builtin(__builtin_amdgcn_s_wait_tensorcnt)
#define HAVE_TDM 1
#else
#define HAVE_TDM 0
#endif

#if HAVE_TDM
// 1-D contiguous copy: n8 elements of 8 bytes, global -> LDS (D# per ISA ch.8)
__device__ __forceinline__ void tdm_load_1d(void* lds, const void* gsrc, int n8) {
    uint32 ldsoff = (uint32)(size_t)(__attribute__((address_space(3))) char*)lds;
    uint64 ga = (uint64)(size_t)gsrc;
    u32x4 g0;
    g0.x = 1u;                                   // count=1, user descriptor
    g0.y = ldsoff;                               // lds_addr (bytes)
    g0.z = (uint32)ga;                           // global_addr[31:0]
    g0.w = (uint32)((ga >> 32) & 0x01FFFFFFu) | 0x80000000u;  // addr hi | type=2
    i32x8 g1;
    g1[0] = 0x00030000;                          // workgroup_mask=0, data_size=8B
    g1[1] = (int)((n8 & 0xFFFF) << 16);          // tensor_dim0[15:0]
    g1[2] = (int)(((uint32)n8 >> 16) & 0xFFFF) | (1 << 16); // dim0 hi | tensor_dim1=1
    g1[3] = (int)((uint32)n8 << 16);             // tile_dim0 (n8 < 65536)
    g1[4] = 0;                                   // tile_dim1/2 unused
    g1[5] = n8;                                  // tensor_dim0_stride lo
    g1[6] = 0;
    g1[7] = 0;
    i32x4 z4 = { 0, 0, 0, 0 };
#if __clang_major__ >= 23
    i32x8 z8 = { 0, 0, 0, 0, 0, 0, 0, 0 };
    __builtin_amdgcn_tensor_load_to_lds(g0, g1, z4, z4, z8, 0);
#else
    __builtin_amdgcn_tensor_load_to_lds(g0, g1, z4, z4, 0);
#endif
}
#endif

// ---------------------------------------------------------------------------
// Conversion kernels (run once)
// ---------------------------------------------------------------------------
__global__ __launch_bounds__(256)
void cvt_act_bf16(const float4* __restrict__ in, uint2* __restrict__ out) {
    size_t i = (size_t)blockIdx.x * 256 + threadIdx.x;
    float4 v = in[i];
    uint2 r;
    r.x = pack_bf2(v.x, v.y);
    r.y = pack_bf2(v.z, v.w);
    out[i] = r;
}

// W f32 [1024][1024] -> bf16 pair-packed: dword[(k>>1)*1024 + n] = {W[k][n], W[k+1][n]}
__global__ __launch_bounds__(256)
void cvt_w_pack(const float* __restrict__ W, uint32* __restrict__ out) {
    size_t idx = (size_t)blockIdx.x * 256 + threadIdx.x;   // 512K dwords
    int n  = (int)(idx & 1023);
    int k  = (int)(idx >> 10) << 1;
    float a = W[(size_t)k * D_MODEL + n];
    float b = W[(size_t)(k + 1) * D_MODEL + n];
    out[idx] = pack_bf2(a, b);
}

// ---------------------------------------------------------------------------
// GEMM: C[4096,1024] = A(bf16 rowmajor) * W(bf16 pair-packed)
// 128 thr / 4 waves; block tile 64x128, BK=32; wave tile 32x64 (8 frags).
// Double-buffered async-LDS staging.
// OUT_MODE 0: f32 [M,N]   1: bf16 head layout [B,H,L,64] (*scale)
// OUT_MODE 2: bf16 key-pair-packed head layout (for V)   (*scale)
// ---------------------------------------------------------------------------
template <int OUT_MODE>
__global__ __launch_bounds__(128)
void gemm_bf16(const bf16* __restrict__ A, const bf16* __restrict__ Wp,
               float* __restrict__ outF, bf16* __restrict__ outH, float scale)
{
    const int n0 = blockIdx.x * 128;
    const int m0 = blockIdx.y * 64;

    __shared__ bf16 As[2][64 * 32];    // rowmajor [m][k]
    __shared__ bf16 Bs[2][16 * 256];   // pair-packed: dword r*128+n, r=k>>1

    const int tid  = threadIdx.x;
    const int lane = tid & 31, wave = tid >> 5;
    const int half = lane >> 4, ml = lane & 15;
    const int wm = (wave >> 1) * 32;
    const int wn = (wave & 1) * 64;

    auto stageA = [&](int buf, int kt) {
        const bf16* src = A + (size_t)m0 * D_MODEL + kt;
        for (int t = tid; t < 256; t += 128) {           // 64 rows x 4 chunks
            int r = t >> 2, s = (t & 3) << 3;
            cp16(src + (size_t)r * D_MODEL + s, &As[buf][r * 32 + s]);
        }
    };
    auto stageB = [&](int buf, int kt) {                 // 16 dw-rows x 128 dw
        const uint32* src = (const uint32*)Wp + (size_t)(kt >> 1) * D_MODEL + n0;
        uint32* dst = (uint32*)Bs[buf];
        for (int t = tid; t < 512; t += 128) {
            int r = t >> 5, s = (t & 31) << 2;
            cp16(src + (size_t)r * D_MODEL + s, dst + r * 128 + s);
        }
    };

    FragF acc[2][4];
    for (int a = 0; a < 2; ++a)
        for (int c = 0; c < 4; ++c)
            for (int i = 0; i < 8; ++i) acc[a][c].f[i] = 0.f;

    stageA(0, 0); stageB(0, 0);
    for (int it = 0; it < D_MODEL / 32; ++it) {
        const int cur = it & 1;
        wait_cp();
        __syncthreads();
        if (it + 1 < D_MODEL / 32) { stageA(cur ^ 1, (it + 1) * 32); stageB(cur ^ 1, (it + 1) * 32); }

        const uint32* Asdw = (const uint32*)As[cur];
        const uint32* Bsdw = (const uint32*)Bs[cur];
        FragBF a0, a1, b[4];
        for (int i = 0; i < 8; ++i) {
            int ka = a_k0(i, half) >> 1;
            a0.u[i] = Asdw[(wm + ml) * 16 + ka];
            a1.u[i] = Asdw[(wm + 16 + ml) * 16 + ka];
            int kb = b_k0(i, half) >> 1;
            b[0].u[i] = Bsdw[kb * 128 + wn + ml];
            b[1].u[i] = Bsdw[kb * 128 + wn + 16 + ml];
            b[2].u[i] = Bsdw[kb * 128 + wn + 32 + ml];
            b[3].u[i] = Bsdw[kb * 128 + wn + 48 + ml];
        }
        for (int c = 0; c < 4; ++c) {
            acc[0][c].v = wmma_bf16(a0, b[c], acc[0][c].v);
            acc[1][c].v = wmma_bf16(a1, b[c], acc[1][c].v);
        }
    }

    for (int sr = 0; sr < 2; ++sr)
        for (int sc = 0; sc < 4; ++sc) {
            FragF* F = &acc[sr][sc];
            int col = n0 + wn + sc * 16 + ml;
            for (int jv = 0; jv < 8; ++jv) {
                int row = m0 + wm + sr * 16 + jv + 8 * half;
                if (OUT_MODE == 0) {
                    outF[(size_t)row * D_MODEL + col] = F->f[jv];
                } else {
                    int b_ = row >> 11, l = row & 2047;
                    int h  = col >> 6,  d = col & 63;
                    size_t bh = (size_t)(b_ * N_HEADS + h);
                    bf16 v = (bf16)(F->f[jv] * scale);
                    if (OUT_MODE == 1) {
                        outH[(bh * L_SEQ + l) * D_HEAD + d] = v;
                    } else {  // key-pair-packed: dword bh*64K + (l>>1)*64 + d
                        size_t dw = bh * (L_SEQ * 32) + (size_t)(l >> 1) * 64 + d;
                        outH[dw * 2 + (l & 1)] = v;
                    }
                }
            }
        }
}

// ---------------------------------------------------------------------------
// Flash attention, 64-key tiles. K/V tiles are contiguous 8KB global images:
// staged by the Tensor Data Mover (one tensor_load_to_lds per tile) when
// available, else async-LDS copies. Double-buffered.
// ---------------------------------------------------------------------------
__global__ __launch_bounds__(128)
void attn_flash(const bf16* __restrict__ Q, const bf16* __restrict__ Kd,
                const bf16* __restrict__ Vp, const float* __restrict__ bias,
                bf16* __restrict__ AO)
{
    const int bh = blockIdx.y;
    const int b  = bh >> 4, h = bh & 15;
    const int tid  = threadIdx.x;
    const int lane = tid & 31, wave = tid >> 5;
    const int half = lane >> 4, ml = lane & 15;
    const int qw   = blockIdx.x * 64 + wave * 16;

    const uint32* qdw = (const uint32*)(Q + (size_t)bh * L_SEQ * D_HEAD);
    const bf16*   kp  = Kd + (size_t)bh * L_SEQ * D_HEAD;
    const bf16*   vpp = Vp + (size_t)bh * L_SEQ * D_HEAD;  // packed, same extent

    __shared__ bf16 Ks[2][64 * 64];    // rowmajor [key][d]      (8KB each)
    __shared__ bf16 Vs[2][64 * 64];    // key-pair-packed image  (8KB each)
    __shared__ bf16 Ps[4][16 * 64];    // per-wave P staging     (2KB each)

    auto stage = [&](int buf, int key0) {
#if HAVE_TDM
        if (tid == 0) {                            // one DMA per tile (EXEC ignored)
            tdm_load_1d(&Ks[buf][0], kp  + (size_t)key0 * 64, 1024);
            tdm_load_1d(&Vs[buf][0], vpp + (size_t)key0 * 64, 1024);
        }
#else
        const bf16* ks = kp  + (size_t)key0 * 64;
        const bf16* vs = vpp + (size_t)key0 * 64;
        for (int t = tid; t < 512; t += 128) {
            cp16(ks + t * 8, &Ks[buf][t * 8]);
            cp16(vs + t * 8, &Vs[buf][t * 8]);
        }
#endif
    };
    auto wait_stage = [&]() {
#if HAVE_TDM
        if (tid < 32) __builtin_amdgcn_s_wait_tensorcnt(0);  // issuing wave only
#else
        wait_cp();
#endif
    };

    FragBF qa0, qa1;
    for (int i = 0; i < 8; ++i) {
        int ka = a_k0(i, half);
        qa0.u[i] = qdw[(size_t)(qw + ml) * 32 + (ka >> 1)];
        qa1.u[i] = qdw[(size_t)(qw + ml) * 32 + ((ka + 32) >> 1)];
    }
    FragBF ones;
    for (int i = 0; i < 8; ++i) ones.u[i] = 0x3F803F80u;   // bf16 1.0 pair

    FragF o[4];
    float mrow[8], lrow[8];
    for (int c = 0; c < 4; ++c)
        for (int i = 0; i < 8; ++i) o[c].f[i] = 0.f;
    for (int i = 0; i < 8; ++i) { mrow[i] = -1e30f; lrow[i] = 0.f; }

    stage(0, 0);
    for (int j = 0; j < L_SEQ / 64; ++j) {
        const int cur = j & 1;
        const int key0 = j * 64;
        wait_stage();
        __syncthreads();
        if (j + 1 < L_SEQ / 64) stage(cur ^ 1, (j + 1) * 64);

        // ---- S = bias + Q K^T : 4 sub-tiles of 16 keys ----
        FragF s[4];
        for (int ss = 0; ss < 4; ++ss)
            for (int jv = 0; jv < 8; ++jv) {
                int qr = qw + jv + 8 * half;
                s[ss].f[jv] = bias[(size_t)qr * L_SEQ + key0 + ss * 16 + ml];
            }
        const uint32* Ksdw = (const uint32*)Ks[cur];
        const uint32* Vsdw = (const uint32*)Vs[cur];
        for (int ss = 0; ss < 4; ++ss) {
            FragBF kb0, kb1;
            for (int i = 0; i < 8; ++i) {
                int kb = b_k0(i, half) >> 1;
                kb0.u[i] = Ksdw[(ss * 16 + ml) * 32 + kb];        // d 0..31
                kb1.u[i] = Ksdw[(ss * 16 + ml) * 32 + 16 + kb];   // d 32..63
            }
            s[ss].v = wmma_bf16(qa0, kb0, s[ss].v);
            s[ss].v = wmma_bf16(qa1, kb1, s[ss].v);
        }

        // ---- running max (one lane-reduction per row per 64 keys) ----
        float alpha[8];
        for (int jv = 0; jv < 8; ++jv) {
            float t = fmaxf(fmaxf(s[0].f[jv], s[1].f[jv]),
                            fmaxf(s[2].f[jv], s[3].f[jv]));
            for (int off = 1; off < 16; off <<= 1) t = fmaxf(t, __shfl_xor(t, off, 32));
            float mnew = fmaxf(mrow[jv], t);
            alpha[jv] = __expf(mrow[jv] - mnew);
            mrow[jv] = mnew;
        }
        // ---- P = exp(S - m); relayout to A-frags via per-wave LDS ----
        bf16* pw = Ps[wave];
        for (int jv = 0; jv < 8; ++jv) {
            int m = (jv + 8 * half) * 64;
            for (int ss = 0; ss < 4; ++ss)
                pw[m + ss * 16 + ml] = (bf16)__expf(s[ss].f[jv] - mrow[jv]);
        }
        FragBF pa0, pa1;
        {
            const uint32* pdw = (const uint32*)pw;   // same-wave LDS: in order
            for (int i = 0; i < 8; ++i) {
                int ka = a_k0(i, half) >> 1;
                pa0.u[i] = pdw[ml * 32 + ka];        // keys 0..31
                pa1.u[i] = pdw[ml * 32 + 16 + ka];   // keys 32..63
            }
        }
        // ---- row sums via WMMA (P @ ones) ----
        FragF rs;
        for (int i = 0; i < 8; ++i) rs.f[i] = 0.f;
        rs.v = wmma_bf16(pa0, ones, rs.v);
        rs.v = wmma_bf16(pa1, ones, rs.v);
        for (int jv = 0; jv < 8; ++jv) lrow[jv] = lrow[jv] * alpha[jv] + rs.f[jv];

        // ---- O = O*alpha + P V (K=64 via two chained WMMAs per d-chunk) ----
        for (int c = 0; c < 4; ++c) {
            for (int jv = 0; jv < 8; ++jv) o[c].f[jv] *= alpha[jv];
            FragBF vb0, vb1;
            for (int i = 0; i < 8; ++i) {
                int kb = b_k0(i, half) >> 1;
                vb0.u[i] = Vsdw[kb * 64 + c * 16 + ml];          // keys 0..31
                vb1.u[i] = Vsdw[(kb + 16) * 64 + c * 16 + ml];   // keys 32..63
            }
            o[c].v = wmma_bf16(pa0, vb0, o[c].v);
            o[c].v = wmma_bf16(pa1, vb1, o[c].v);
        }
    }

    // epilogue: normalize, recombine heads -> AO[b*L + qr][h*64 + d] (bf16)
    for (int c = 0; c < 4; ++c)
        for (int jv = 0; jv < 8; ++jv) {
            int qr  = qw + jv + 8 * half;
            int col = h * D_HEAD + c * 16 + ml;
            AO[(size_t)(b * L_SEQ + qr) * D_MODEL + col] = (bf16)(o[c].f[jv] / lrow[jv]);
        }
}

// ---------------------------------------------------------------------------
extern "C" void kernel_launch(void* const* d_in, const int* in_sizes, int n_in,
                              void* d_out, int out_size, void* d_ws, size_t ws_size,
                              hipStream_t stream) {
    const float* xq   = (const float*)d_in[0];
    const float* xm   = (const float*)d_in[1];
    const float* bias = (const float*)d_in[2];
    const float* Wq   = (const float*)d_in[3];
    const float* Wk   = (const float*)d_in[4];
    const float* Wv   = (const float*)d_in[5];
    const float* Wo   = (const float*)d_in[6];
    float* out = (float*)d_out;

    const size_t EA = (size_t)BL * D_MODEL;       // 4M elems (activations)
    const size_t EW = (size_t)D_MODEL * D_MODEL;  // 1M elems (weights)
    bf16* xqb = (bf16*)d_ws;
    bf16* xmb = xqb + EA;
    bf16* wqp = xmb + EA;
    bf16* wkp = wqp + EW;
    bf16* wvp = wkp + EW;
    bf16* wop = wvp + EW;
    bf16* qb  = wop + EW;
    bf16* kb  = qb  + EA;
    bf16* vb  = kb  + EA;
    bf16* ao  = vb  + EA;

    // 1) one-time conversions / packing
    cvt_act_bf16<<<EA / 1024, 256, 0, stream>>>((const float4*)xq, (uint2*)xqb);
    cvt_act_bf16<<<EA / 1024, 256, 0, stream>>>((const float4*)xm, (uint2*)xmb);
    cvt_w_pack<<<EW / 512, 256, 0, stream>>>(Wq, (uint32*)wqp);
    cvt_w_pack<<<EW / 512, 256, 0, stream>>>(Wk, (uint32*)wkp);
    cvt_w_pack<<<EW / 512, 256, 0, stream>>>(Wv, (uint32*)wvp);
    cvt_w_pack<<<EW / 512, 256, 0, stream>>>(Wo, (uint32*)wop);

    // 2) projections (Q carries d_head^-0.5); V stored key-pair-packed
    dim3 gG(D_MODEL / 128, BL / 64), blk(128);
    gemm_bf16<1><<<gG, blk, 0, stream>>>(xqb, wqp, nullptr, qb, 0.125f);
    gemm_bf16<1><<<gG, blk, 0, stream>>>(xmb, wkp, nullptr, kb, 1.0f);
    gemm_bf16<2><<<gG, blk, 0, stream>>>(xmb, wvp, nullptr, vb, 1.0f);

    // 3) attention
    dim3 gA(L_SEQ / 64, 2 * N_HEADS);
    attn_flash<<<gA, blk, 0, stream>>>(qb, kb, vb, bias, ao);

    // 4) output projection -> f32
    gemm_bf16<0><<<gG, blk, 0, stream>>>(ao, wop, out, nullptr, 1.0f);
}
